// DeepSeekMoE_63324997812260
// MI455X (gfx1250) — matrix operand order repaired and checked by reading, not compiled, not run on hardware
//
#include <hip/hip_runtime.h>
#include <hip/hip_bf16.h>
#include <math.h>

// ---------------------------------------------------------------------------
// DeepSeek MoE on gfx1250 (MI455X): sparse top-2 grouped GEMM, bf16 WMMA with
// f32 accumulation, in-register fp32->bf16 conversion of the weight stream.
// HBM floor: ~800 MB fp32 weights once -> ~34 us @ 23.3 TB/s. M-tile = 64
// tokens so each staged weight tile is reused 4x16 rows -> L2 re-read traffic
// (~1.5 GB) stays below the HBM floor; sparse compute (~48 GFLOP bf16) is
// negligible at WMMA rates.
// ---------------------------------------------------------------------------

typedef __attribute__((ext_vector_type(16))) __bf16 v16bf;
typedef __attribute__((ext_vector_type(8)))  __bf16 v8bf;
typedef __attribute__((ext_vector_type(8)))  float  v8f;
typedef int v4i __attribute__((vector_size(16)));   // matches async-LDS builtin

#define D_DIM 5120
#define F_DIM 384
#define E_RT  32
#define E_TOT 34      // 32 routed + 2 shared
#define T_TOK 1024
#define KC    64      // K-chunk staged in LDS per iteration
#define LDK   72      // padded LDS row stride (bf16 elems)
#define MT    64      // token rows per block (4 x 16-row WMMA sub-tiles)

// Workspace byte offsets
#define WS_COMBINE 0u
#define WS_TOKID   131072u
#define WS_TOKW    262144u
#define WS_COUNTS  393216u
#define WS_ABUF    524288u    // 34*1024*384 bf16

#define AS1 __attribute__((address_space(1)))
#define AS3 __attribute__((address_space(3)))

#if defined(__has_builtin)
#if __has_builtin(__builtin_amdgcn_global_load_async_to_lds_b128)
#define USE_ASYNC_LDS 1
#endif
#endif

__device__ __forceinline__ void wait_async0() {
#if defined(__has_builtin) && __has_builtin(__builtin_amdgcn_s_wait_asynccnt)
  __builtin_amdgcn_s_wait_asynccnt(0);
#else
  asm volatile("s_wait_asynccnt 0x0" ::: "memory");
#endif
}

// pack two f32 -> one dword of two bf16 (lowers to v_cvt_pk_bf16_f32)
__device__ __forceinline__ unsigned pack_bf16(float a, float b) {
  union { unsigned u; __bf16 h[2]; } p;
  p.h[0] = (__bf16)a; p.h[1] = (__bf16)b;
  return p.u;
}

// ---- WMMA fragment loads from LDS (row-major [row][k], stride LDK) ---------
// A (16-bit, 16x32): lanes 0-15 -> M, VGPR0-3 K = khalf*8 + 0..7, VGPR4-7 += 16
__device__ __forceinline__ v16bf frag_a(const __bf16* row, int ks, int lane) {
  const int kb = ks + ((lane >> 4) << 3);
  v8bf lo = *(const v8bf*)(row + kb);
  v8bf hi = *(const v8bf*)(row + kb + 16);
  return __builtin_shufflevector(lo, hi, 0,1,2,3,4,5,6,7,8,9,10,11,12,13,14,15);
}
// B (16-bit, 32x16): lanes 0-15 -> N, VGPR0-3 K = khalf*8 + 0..7, VGPR4-7 += 16
__device__ __forceinline__ v16bf frag_b(const __bf16* row, int ks, int lane) {
  const int kb = ks + ((lane >> 4) << 3);
  v8bf lo = *(const v8bf*)(row + kb);
  v8bf hi = *(const v8bf*)(row + kb + 16);
  return __builtin_shufflevector(lo, hi, 0,1,2,3,4,5,6,7,8,9,10,11,12,13,14,15);
}

// ---- Kernel 1: gating (one wave32 per token) -------------------------------
__global__ __launch_bounds__(32)
void moe_gate_kernel(const float* __restrict__ x, const float* __restrict__ gw,
                     const float* __restrict__ gb, float* __restrict__ combine)
{
  const int t = blockIdx.x;
  const int e = threadIdx.x;
  const float* xr = x + (size_t)t * D_DIM;
  float acc = gb[e];
  for (int d = 0; d < D_DIM; d += 4) {
    acc += xr[d]   * gw[(d    ) * E_RT + e];
    acc += xr[d+1] * gw[(d + 1) * E_RT + e];
    acc += xr[d+2] * gw[(d + 2) * E_RT + e];
    acc += xr[d+3] * gw[(d + 3) * E_RT + e];
  }
  float m = acc;
  for (int o = 16; o > 0; o >>= 1) m = fmaxf(m, __shfl_xor(m, o, 32));
  float p = __expf(acc - m);
  float s = p;
  for (int o = 16; o > 0; o >>= 1) s += __shfl_xor(s, o, 32);
  p /= s;
  float v1 = p; int i1 = e;
  for (int o = 16; o > 0; o >>= 1) {
    float ov = __shfl_xor(v1, o, 32); int oi = __shfl_xor(i1, o, 32);
    if (ov > v1 || (ov == v1 && oi < i1)) { v1 = ov; i1 = oi; }
  }
  float v2 = (e == i1) ? -1.f : p; int i2 = e;
  for (int o = 16; o > 0; o >>= 1) {
    float ov = __shfl_xor(v2, o, 32); int oi = __shfl_xor(i2, o, 32);
    if (ov > v2 || (ov == v2 && oi < i2)) { v2 = ov; i2 = oi; }
  }
  combine[t * E_RT + e] = (e == i1 || e == i2) ? p : 0.f;
}

// ---- Kernel 2: deterministic per-expert token list compaction --------------
__global__ __launch_bounds__(32)
void moe_build_lists_kernel(const float* __restrict__ combine,
                            int* __restrict__ tokid, float* __restrict__ tokw,
                            int* __restrict__ counts)
{
  const int e = blockIdx.x;
  const int lane = threadIdx.x;
  int base = 0;
  for (int i = 0; i < T_TOK / 32; ++i) {
    const int t = i * 32 + lane;
    const float w = combine[t * E_RT + e];
    const bool pred = (w > 0.f);
    const unsigned m32 = (unsigned)__ballot(pred);
    const int off = __popc(m32 & ((1u << lane) - 1u));
    if (pred) {
      tokid[e * T_TOK + base + off] = t;
      tokw [e * T_TOK + base + off] = w;
    }
    base += __popc(m32);
  }
  if (lane == 0) counts[e] = base;
}

// ---- Kernel 3: out = shared_b2.sum(0) --------------------------------------
__global__ __launch_bounds__(256)
void moe_init_out_kernel(float* __restrict__ out, const float* __restrict__ sb2)
{
  const int i = blockIdx.x * 256 + threadIdx.x;
  const int d = i % D_DIM;
  out[i] = sb2[d] + sb2[D_DIM + d];
}

// ---- Kernel 4: fc1 + SwiGLU (64 tok x (64 v + 64 g) cols per block) --------
__global__ __launch_bounds__(256)
void moe_fc1_kernel(const float* __restrict__ x,
                    const float* __restrict__ rw1, const float* __restrict__ rb1,
                    const float* __restrict__ sw1, const float* __restrict__ sb1,
                    const int* __restrict__ tokid, const float* __restrict__ tokw,
                    const int* __restrict__ counts, __bf16* __restrict__ abuf)
{
  __shared__ union {
    struct { __bf16 a[MT * LDK]; __bf16 b[128 * LDK]; } st;  // K-loop staging
    float c[MT * 132];                                       // post-loop C xchg
  } u;
  __shared__ int   sTid[MT];
  __shared__ float sWt[MT];

  const int e   = blockIdx.z;
  const int f0  = blockIdx.y * 64;
  const int tt  = blockIdx.x;
  const int tid = threadIdx.x;
  const int lane = tid & 31, w = tid >> 5;

  const int cnt = (e < E_RT) ? counts[e] : T_TOK;
  if (tt * MT >= cnt) return;

  const float* w1 = (e < E_RT) ? (rw1 + (size_t)e * D_DIM * (2 * F_DIM))
                               : (sw1 + (size_t)(e - E_RT) * D_DIM * (2 * F_DIM));
  const float* b1 = (e < E_RT) ? (rb1 + e * (2 * F_DIM))
                               : (sb1 + (e - E_RT) * (2 * F_DIM));

  if (tid < MT) {
    const int slot = tt * MT + tid;
    int t = -1; float wt = 0.f;
    if (slot < cnt) {
      if (e < E_RT) { t = tokid[e * T_TOK + slot]; wt = tokw[e * T_TOK + slot]; }
      else          { t = slot;                    wt = 1.f; }
    }
    sTid[tid] = t; sWt[tid] = wt;
  }
  __syncthreads();

  // Branch-free hoisted A-loader state: 4 rows per thread (invalid rows clamp
  // to token 0; their C rows are discarded by slot<cnt guards).
  const int arl = tid >> 4;                // base A row (0..15)
  const int ak  = (tid & 15) * 4;          // float4 k offset
  const float* arowp[4];
  #pragma unroll
  for (int it = 0; it < 4; ++it) {
    const int t0 = sTid[it * 16 + arl];
    arowp[it] = x + (size_t)(t0 < 0 ? 0 : t0) * D_DIM + ak;
  }

  const int n4 = lane * 4;                 // B: 4 consecutive n per thread
  const int col = (n4 < 64) ? (f0 + n4) : (F_DIM + f0 + (n4 - 64));
  const float* bbase = w1 + col;
  const int kw = w * 2;                    // wave-local k pair base

  v8f acc[4] = {};
  for (int k0 = 0; k0 < D_DIM; k0 += KC) {
    __syncthreads();
    // A: 64 x 64 fp32 -> bf16, four float4 per thread
    #pragma unroll
    for (int it = 0; it < 4; ++it) {
      const float4 v = *(const float4*)(arowp[it] + k0);
      uint2 pk; pk.x = pack_bf16(v.x, v.y); pk.y = pack_bf16(v.z, v.w);
      *(uint2*)(u.st.a + (it * 16 + arl) * LDK + ak) = pk;
    }
    // B: 128 n x 64 k, paired k rows -> packed bf16 dwords
    #pragma unroll
    for (int it = 0; it < 4; ++it) {
      const int kk = it * 16 + kw;
      const float* bp = bbase + (size_t)(k0 + kk) * (2 * F_DIM);
      const float4 r0 = *(const float4*)bp;
      const float4 r1 = *(const float4*)(bp + 2 * F_DIM);
      if (it == 0 && k0 + KC < D_DIM)
        __builtin_prefetch(bp + (size_t)KC * (2 * F_DIM), 0, 3);
      *(unsigned*)(u.st.b + (n4 + 0) * LDK + kk) = pack_bf16(r0.x, r1.x);
      *(unsigned*)(u.st.b + (n4 + 1) * LDK + kk) = pack_bf16(r0.y, r1.y);
      *(unsigned*)(u.st.b + (n4 + 2) * LDK + kk) = pack_bf16(r0.z, r1.z);
      *(unsigned*)(u.st.b + (n4 + 3) * LDK + kk) = pack_bf16(r0.w, r1.w);
    }
    __syncthreads();
    const __bf16* br = u.st.b + (w * 16 + (lane & 15)) * LDK;
    #pragma unroll
    for (int ks = 0; ks < KC; ks += 32) {
      const v16bf b = frag_b(br, ks, lane);
      #pragma unroll
      for (int mt = 0; mt < 4; ++mt) {
        const v16bf a = frag_a(u.st.a + (mt * 16 + (lane & 15)) * LDK, ks, lane);
        acc[mt] = __builtin_amdgcn_wmma_f32_16x16x32_bf16(
            false, a, false, b, (short)0, acc[mt], false, false);
      }
    }
  }
  __syncthreads();
  // bias, stash f32 tiles so v/g column pairs can meet for SwiGLU
  {
    const int nblk = w * 16 + (lane & 15);
    const int colg = (nblk < 64) ? (f0 + nblk) : (F_DIM + f0 + (nblk - 64));
    const float bias = b1[colg];
    const int mb = (lane >> 4) * 8;
    #pragma unroll
    for (int mt = 0; mt < 4; ++mt)
      #pragma unroll
      for (int i = 0; i < 8; ++i)
        u.c[(mt * 16 + mb + i) * 132 + nblk] = acc[mt][i] + bias;
  }
  __syncthreads();
  // SwiGLU: silu(v) * g * combine_weight -> compact bf16 activation buffer
  #pragma unroll
  for (int it = 0; it < 16; ++it) {
    const int idx = it * 256 + tid;
    const int m = idx >> 6, f = idx & 63;
    const int slot = tt * MT + m;
    if (slot < cnt) {
      const float v = u.c[m * 132 + f];
      const float g = u.c[m * 132 + 64 + f];
      const float a = (v / (1.f + __expf(-v))) * g * sWt[m];
      abuf[((size_t)e * T_TOK + slot) * F_DIM + f0 + f] = (__bf16)a;
    }
  }
}

// ---- Kernel 5: fc2 + weighted bias + scatter-accumulate --------------------
__global__ __launch_bounds__(256)
void moe_fc2_kernel(const __bf16* __restrict__ abuf,
                    const float* __restrict__ rw2, const float* __restrict__ rb2,
                    const float* __restrict__ sw2,
                    const int* __restrict__ tokid, const float* __restrict__ tokw,
                    const int* __restrict__ counts, float* __restrict__ out)
{
  __shared__ union {
    struct { __bf16 a[MT * LDK]; __bf16 b[128 * LDK]; } st;
  } u;
  __shared__ int   sTid[MT];
  __shared__ float sWt[MT];

  const int e   = blockIdx.z;
  const int d0  = blockIdx.y * 128;
  const int tt  = blockIdx.x;
  const int tid = threadIdx.x;
  const int lane = tid & 31, w = tid >> 5;

  const int cnt = (e < E_RT) ? counts[e] : T_TOK;
  if (tt * MT >= cnt) return;

  const float* w2 = (e < E_RT) ? (rw2 + (size_t)e * F_DIM * D_DIM)
                               : (sw2 + (size_t)(e - E_RT) * F_DIM * D_DIM);

  if (tid < MT) {
    const int slot = tt * MT + tid;
    int t = -1; float wt = 0.f;
    if (slot < cnt) {
      if (e < E_RT) { t = tokid[e * T_TOK + slot]; wt = tokw[e * T_TOK + slot]; }
      else          { t = slot;                    wt = 1.f; }
    }
    sTid[tid] = t; sWt[tid] = wt;
  }
  __syncthreads();

  // A staging: abuf is already bf16. Async global->LDS (ASYNCcnt path) when
  // available: each thread moves 2x16B of its row; else plain vector loads.
  const int ar8  = tid >> 2;                 // row 0..63
  const int seg2 = (tid & 3) * 16;           // bf16 elems: {0,16,32,48}
  int slc = tt * MT + ar8; if (slc >= cnt) slc = cnt - 1;
  const __bf16* agsrc = abuf + ((size_t)e * T_TOK + slc) * F_DIM + seg2;
  __bf16* aldst = u.st.a + ar8 * LDK + seg2;

  const int n4 = lane * 4;
  const float* bbase = w2 + (d0 + n4);
  const int kw = w * 2;

  v8f acc[4] = {};
  for (int k0 = 0; k0 < F_DIM; k0 += KC) {
    __syncthreads();
#ifdef USE_ASYNC_LDS
    __builtin_amdgcn_global_load_async_to_lds_b128(
        (AS1 v4i*)(uintptr_t)(const void*)(agsrc + k0),
        (AS3 v4i*)(uintptr_t)(void*)aldst, 0, 0);
    __builtin_amdgcn_global_load_async_to_lds_b128(
        (AS1 v4i*)(uintptr_t)(const void*)(agsrc + k0),
        (AS3 v4i*)(uintptr_t)(void*)aldst, 16, 0);
#else
    *(uint4*)aldst        = *(const uint4*)(agsrc + k0);
    *(uint4*)(aldst + 8)  = *(const uint4*)(agsrc + k0 + 8);
#endif
    #pragma unroll
    for (int it = 0; it < 4; ++it) {
      const int kk = it * 16 + kw;
      const float* bp = bbase + (size_t)(k0 + kk) * D_DIM;
      const float4 r0 = *(const float4*)bp;
      const float4 r1 = *(const float4*)(bp + D_DIM);
      if (it == 0 && k0 + KC < F_DIM)
        __builtin_prefetch(bp + (size_t)KC * D_DIM, 0, 3);
      *(unsigned*)(u.st.b + (n4 + 0) * LDK + kk) = pack_bf16(r0.x, r1.x);
      *(unsigned*)(u.st.b + (n4 + 1) * LDK + kk) = pack_bf16(r0.y, r1.y);
      *(unsigned*)(u.st.b + (n4 + 2) * LDK + kk) = pack_bf16(r0.z, r1.z);
      *(unsigned*)(u.st.b + (n4 + 3) * LDK + kk) = pack_bf16(r0.w, r1.w);
    }
#ifdef USE_ASYNC_LDS
    wait_async0();
#endif
    __syncthreads();
    const __bf16* br = u.st.b + (w * 16 + (lane & 15)) * LDK;
    #pragma unroll
    for (int ks = 0; ks < KC; ks += 32) {
      const v16bf b = frag_b(br, ks, lane);
      #pragma unroll
      for (int mt = 0; mt < 4; ++mt) {
        const v16bf a = frag_a(u.st.a + (mt * 16 + (lane & 15)) * LDK, ks, lane);
        acc[mt] = __builtin_amdgcn_wmma_f32_16x16x32_bf16(
            false, a, false, b, (short)0, acc[mt], false, false);
      }
    }
  }
  // scatter: out[t, d] += acc + combine*routed_b2 (shared bias handled by init)
  const int n = lane & 15;
  const int d = d0 + w * 16 + n;
  const int mb = (lane >> 4) * 8;
  const bool routed = (e < E_RT);
  const float bias_col = routed ? rb2[e * D_DIM + d] : 0.f;
  #pragma unroll
  for (int mt = 0; mt < 4; ++mt) {
    #pragma unroll
    for (int i = 0; i < 8; ++i) {
      const int m = mt * 16 + mb + i;
      const int slot = tt * MT + m;
      if (slot < cnt) {
        const int t = sTid[m];
        const float val = acc[mt][i] + (routed ? sWt[m] * bias_col : 0.f);
        unsafeAtomicAdd(&out[(size_t)t * D_DIM + d], val);
      }
    }
  }
}

// ---------------------------------------------------------------------------
extern "C" void kernel_launch(void* const* d_in, const int* in_sizes, int n_in,
                              void* d_out, int out_size, void* d_ws, size_t ws_size,
                              hipStream_t stream)
{
  (void)in_sizes; (void)n_in; (void)out_size; (void)ws_size;
  const float* x   = (const float*)d_in[0];
  const float* gw  = (const float*)d_in[1];
  const float* gb  = (const float*)d_in[2];
  const float* sw1 = (const float*)d_in[3];
  const float* sb1 = (const float*)d_in[4];
  const float* sw2 = (const float*)d_in[5];
  const float* sb2 = (const float*)d_in[6];
  const float* rw1 = (const float*)d_in[7];
  const float* rb1 = (const float*)d_in[8];
  const float* rw2 = (const float*)d_in[9];
  const float* rb2 = (const float*)d_in[10];
  float* out = (float*)d_out;

  char* ws = (char*)d_ws;
  float*  combine = (float*)(ws + WS_COMBINE);
  int*    tokid   = (int*)  (ws + WS_TOKID);
  float*  tokw    = (float*)(ws + WS_TOKW);
  int*    counts  = (int*)  (ws + WS_COUNTS);
  __bf16* abuf    = (__bf16*)(ws + WS_ABUF);

  moe_gate_kernel<<<T_TOK, 32, 0, stream>>>(x, gw, gb, combine);
  moe_build_lists_kernel<<<E_RT, 32, 0, stream>>>(combine, tokid, tokw, counts);
  moe_init_out_kernel<<<(T_TOK * D_DIM) / 256, 256, 0, stream>>>(out, sb2);
  moe_fc1_kernel<<<dim3(T_TOK / MT, F_DIM / 64, E_TOT), 256, 0, stream>>>(
      x, rw1, rb1, sw1, sb1, tokid, tokw, counts, abuf);
  moe_fc2_kernel<<<dim3(T_TOK / MT, D_DIM / 128, E_TOT), 256, 0, stream>>>(
      abuf, rw2, rb2, sw2, tokid, tokw, counts, out);
}
